// CausalSelfAttention_61598420959846
// MI455X (gfx1250) — matrix operand order repaired and checked by reading, not compiled, not run on hardware
//
#include <hip/hip_runtime.h>
#include <hip/hip_bf16.h>

// ---- CDNA5 WMMA vector types ----
typedef __attribute__((ext_vector_type(16))) __bf16 v16bf;
typedef __attribute__((ext_vector_type(8)))  __bf16 v8bf;
typedef __attribute__((ext_vector_type(8)))  float  v8f;

#define B_SZ 4
#define T_SZ 2048
#define C_SZ 2048
#define H_SZ 16
#define D_SZ 128
#define QKV_ROW 6144          // 3*C
#define ATTN_SCALE 0.08838834764831845f  // 1/sqrt(128)

// ---------------------------------------------------------------------------
// fp32 -> bf16 conversion
// ---------------------------------------------------------------------------
__global__ __launch_bounds__(256) void f32_to_bf16_kernel(
    const float* __restrict__ in, __bf16* __restrict__ out, int n) {
  int i = blockIdx.x * blockDim.x + threadIdx.x;
  int stride = gridDim.x * blockDim.x;
  for (; i < n; i += stride) out[i] = (__bf16)in[i];
}

// ---------------------------------------------------------------------------
// GEMM: out[m,n] = sum_k A[m,k] * W[n,k]   (A: MxK bf16, W: NxK bf16)
// One wave computes a 32(M) x 64(N) tile; 8 WMMAs per K=32 step.
// Ping-pong fragment buffers with a 2x-unrolled K loop: buffer P loads while
// buffer Q's WMMA burst executes, with no cross-buffer register copies
// (avoids v_mov + WAR-hazard v_nop padding between WMMAs).
// ---------------------------------------------------------------------------
__global__ __launch_bounds__(256) void gemm_bf16_wmma(
    const __bf16* __restrict__ A, const __bf16* __restrict__ W,
    float* __restrict__ outf, __bf16* __restrict__ outb,
    int M, int N, int K, int n64tiles) {
  const int lane = threadIdx.x & 31;
  const int widx = threadIdx.x >> 5;
  const int gw = blockIdx.x * 8 + widx;
  const int mt = gw / n64tiles;
  const int nt = gw % n64tiles;
  const int mbase = mt * 32;
  const int nbase = nt * 64;
  if (mbase >= M) return;                 // wave-uniform: EXEC stays all-1s
  const int hi  = lane >> 4;              // which 16-lane half
  const int l15 = lane & 15;

  const __bf16* arow = A + (size_t)(mbase + l15) * K + 8 * hi;   // M rows i*16
  const __bf16* wrow = W + (size_t)(nbase + l15) * K + 8 * hi;   // N cols f*16

  // A fragment (16x32 bf16): lanes 0-15 hold M=lane; elems 0..7 = K k..k+7,
  // elems 8..15 = K k+16..k+23; hi half shifted by +8 in K.
  // B fragment (32x16): lane n holds column n; elems 0..7 = K k..k+7,
  // elems 8..15 = K k+16..k+23; hi half shifted by +8 in K (same as A).
  auto load_frags = [&](int k, v16bf Af[2], v16bf Bf[4]) {
#pragma unroll
    for (int i = 0; i < 2; ++i) {
      const __bf16* p = arow + (size_t)i * 16 * K + k;
      v8bf lo = *(const v8bf*)p;
      v8bf hv = *(const v8bf*)(p + 16);
#pragma unroll
      for (int e = 0; e < 8; ++e) { Af[i][e] = lo[e]; Af[i][8 + e] = hv[e]; }
    }
#pragma unroll
    for (int f = 0; f < 4; ++f) {
      const __bf16* p = wrow + (size_t)f * 16 * K + k;
      v8bf b0 = *(const v8bf*)p;
      v8bf b1 = *(const v8bf*)(p + 16);
#pragma unroll
      for (int e = 0; e < 8; ++e) { Bf[f][e] = b0[e]; Bf[f][8 + e] = b1[e]; }
    }
  };

  v8f zero = {};
  v8f acc[2][4];
#pragma unroll
  for (int i = 0; i < 2; ++i)
#pragma unroll
    for (int f = 0; f < 4; ++f) acc[i][f] = zero;

  v16bf A0[2], B0[4], A1[2], B1[4];
  load_frags(0, A0, B0);

  // K is a multiple of 64 for every call (2048).
  for (int k = 0; k < K; k += 64) {
    load_frags(k + 32, A1, B1);           // prefetch odd step
#pragma unroll
    for (int i = 0; i < 2; ++i)
#pragma unroll
      for (int f = 0; f < 4; ++f)
        acc[i][f] = __builtin_amdgcn_wmma_f32_16x16x32_bf16(
            false, A0[i], false, B0[f], (short)0, acc[i][f], false, false);
    if (k + 64 < K) load_frags(k + 64, A0, B0);  // prefetch next even step
#pragma unroll
    for (int i = 0; i < 2; ++i)
#pragma unroll
      for (int f = 0; f < 4; ++f)
        acc[i][f] = __builtin_amdgcn_wmma_f32_16x16x32_bf16(
            false, A1[i], false, B1[f], (short)0, acc[i][f], false, false);
  }

  // C layout: elem r -> row = r + 8*hi, col = lane&15 (per 16-wide n-tile)
#pragma unroll
  for (int i = 0; i < 2; ++i) {
#pragma unroll
    for (int f = 0; f < 4; ++f) {
#pragma unroll
      for (int r = 0; r < 8; ++r) {
        int row = mbase + i * 16 + r + 8 * hi;
        int col = nbase + f * 16 + l15;
        float v = acc[i][f][r];
        if (outf) outf[(size_t)row * N + col] = v;
        else      outb[(size_t)row * N + col] = (__bf16)v;
      }
    }
  }
}

// ---------------------------------------------------------------------------
// Flash attention: one wave per (b, h, 16-row q tile). Online softmax,
// 32-key blocks, causal mask. qkv layout: [b][t][3C] with q|k|v at
// offsets 0 / 2048 / 4096, head h at +h*128 (d contiguous).
// ---------------------------------------------------------------------------
#define AT_WAVES 4
__global__ __launch_bounds__(128) void attn_flash_wmma(
    const __bf16* __restrict__ qkv, __bf16* __restrict__ attn_out) {
  __shared__ __align__(16) __bf16 ldsV[AT_WAVES][32 * 128];  // 8KB/wave
  __shared__ __align__(16) __bf16 ldsP[AT_WAVES][16 * 32];   // 1KB/wave

  const int lane = threadIdx.x & 31;
  const int w    = threadIdx.x >> 5;
  const int gw   = blockIdx.x * AT_WAVES + w;
  const int qt   = gw & 127;          // T/16 = 128 q tiles
  const int h    = (gw >> 7) & 15;
  const int b    = gw >> 11;
  const int qbase = qt * 16;
  const int hi  = lane >> 4;
  const int l15 = lane & 15;

  const __bf16* base = qkv + (size_t)b * T_SZ * QKV_ROW;

  // ---- Q fragments: 4 x (16x32 bf16) covering d = 0..127 ----
  v16bf Qf[4];
  {
    const __bf16* qrow = base + (size_t)(qbase + l15) * QKV_ROW + h * D_SZ;
#pragma unroll
    for (int c = 0; c < 4; ++c) {
      v8bf lo = *(const v8bf*)(qrow + c * 32 + 8 * hi);
      v8bf hv = *(const v8bf*)(qrow + c * 32 + 16 + 8 * hi);
#pragma unroll
      for (int e = 0; e < 8; ++e) { Qf[c][e] = lo[e]; Qf[c][8 + e] = hv[e]; }
    }
  }

  v8f zero = {};
  v8f O[8];
#pragma unroll
  for (int dt = 0; dt < 8; ++dt) O[dt] = zero;
  float mrow[8], lrow[8];
#pragma unroll
  for (int r = 0; r < 8; ++r) { mrow[r] = -3.0e38f; lrow[r] = 0.0f; }

  for (int kb = 0; kb < qbase + 16; kb += 32) {
    // ---- stage V tile (32 keys x 128 d) into per-wave LDS ----
    {
      const uint4* src = (const uint4*)(base + (size_t)(kb + lane) * QKV_ROW +
                                        2 * C_SZ + h * D_SZ);
      uint4* dst = (uint4*)(&ldsV[w][lane * 128]);
#pragma unroll
      for (int i = 0; i < 16; ++i) dst[i] = src[i];
    }

    // ---- load all 8 K fragments first (one load clause), then WMMA burst ----
    v16bf Kf[2][4];
#pragma unroll
    for (int nt2 = 0; nt2 < 2; ++nt2) {
      const __bf16* krow = base + (size_t)(kb + nt2 * 16 + l15) * QKV_ROW +
                           C_SZ + h * D_SZ;
#pragma unroll
      for (int c = 0; c < 4; ++c) {
        v8bf b0 = *(const v8bf*)(krow + c * 32 + 8 * hi);
        v8bf b1 = *(const v8bf*)(krow + c * 32 + 8 * hi + 16);
#pragma unroll
        for (int e = 0; e < 8; ++e) { Kf[nt2][c][e] = b0[e]; Kf[nt2][c][8 + e] = b1[e]; }
      }
    }

    // ---- S = Q K^T for two 16-key tiles (8 WMMAs) ----
    v8f S[2];
#pragma unroll
    for (int nt2 = 0; nt2 < 2; ++nt2) {
      v8f s = zero;
#pragma unroll
      for (int c = 0; c < 4; ++c)
        s = __builtin_amdgcn_wmma_f32_16x16x32_bf16(
            false, Qf[c], false, Kf[nt2][c], (short)0, s, false, false);
      S[nt2] = s;
    }

    // ---- causal mask + online softmax (row = r + 8*hi, col = lane&15) ----
#pragma unroll
    for (int r = 0; r < 8; ++r) {
      int row = qbase + r + 8 * hi;
      float s0 = S[0][r] * ATTN_SCALE; if (kb + l15 > row)      s0 = -1.0e30f;
      float s1 = S[1][r] * ATTN_SCALE; if (kb + 16 + l15 > row) s1 = -1.0e30f;
      float mx = fmaxf(s0, s1);
#pragma unroll
      for (int off = 1; off < 16; off <<= 1)
        mx = fmaxf(mx, __shfl_xor(mx, off, 32));   // stays within 16-lane half
      float mnew  = fmaxf(mrow[r], mx);
      float alpha = __expf(mrow[r] - mnew);
      float p0 = __expf(s0 - mnew);
      float p1 = __expf(s1 - mnew);
      float sum = p0 + p1;
#pragma unroll
      for (int off = 1; off < 16; off <<= 1)
        sum += __shfl_xor(sum, off, 32);
      lrow[r] = lrow[r] * alpha + sum;
      mrow[r] = mnew;
#pragma unroll
      for (int dt = 0; dt < 8; ++dt) O[dt][r] = O[dt][r] * alpha;
      ldsP[w][(r + 8 * hi) * 32 + l15]      = (__bf16)p0;
      ldsP[w][(r + 8 * hi) * 32 + 16 + l15] = (__bf16)p1;
    }

    // ---- re-shape P: C-layout (LDS) -> A-layout fragment (16x32) ----
    v16bf Pa;
#pragma unroll
    for (int e = 0; e < 16; ++e) {
      int key = ((e & 8) ? 16 : 0) + 8 * hi + (e & 7);
      Pa[e] = ldsP[w][l15 * 32 + key];
    }

    // ---- O += P @ V (8 WMMAs over d tiles) ----
#pragma unroll
    for (int dt = 0; dt < 8; ++dt) {
      v16bf Vf;
      int dcol = dt * 16 + l15;
#pragma unroll
      for (int e = 0; e < 16; ++e) {
        int key = ((e & 8) ? 16 : 0) + 8 * hi + (e & 7);
        Vf[e] = ldsV[w][key * 128 + dcol];
      }
      O[dt] = __builtin_amdgcn_wmma_f32_16x16x32_bf16(
          false, Pa, false, Vf, (short)0, O[dt], false, false);
    }
  }

  // ---- normalize and store [B,T,H*D] bf16 ----
#pragma unroll
  for (int r = 0; r < 8; ++r) {
    float inv = 1.0f / lrow[r];
    int row = qbase + r + 8 * hi;
    __bf16* orow = attn_out + ((size_t)(b * T_SZ + row)) * C_SZ + h * D_SZ;
#pragma unroll
    for (int dt = 0; dt < 8; ++dt)
      orow[dt * 16 + l15] = (__bf16)(O[dt][r] * inv);
  }
}

// ---------------------------------------------------------------------------
// Launch: convert -> QKV GEMM -> flash attention -> output GEMM
// Workspace layout (bytes):
//   [0,   32MB)  x bf16            (8192 x 2048)
//   [32,  56MB)  W_qkv bf16        (6144 x 2048)
//   [56,  64MB)  W_out bf16        (2048 x 2048)
//   [64, 160MB)  qkv bf16          (8192 x 6144)
//   [160,192MB)  attn_out bf16     (8192 x 2048)
// ---------------------------------------------------------------------------
extern "C" void kernel_launch(void* const* d_in, const int* in_sizes, int n_in,
                              void* d_out, int out_size, void* d_ws, size_t ws_size,
                              hipStream_t stream) {
  const float* x    = (const float*)d_in[0];
  const float* Wqkv = (const float*)d_in[1];
  const float* Wout = (const float*)d_in[2];
  float* out = (float*)d_out;

  char* ws = (char*)d_ws;
  __bf16* xb    = (__bf16*)(ws);
  __bf16* wqkvb = (__bf16*)(ws + (size_t)33554432);
  __bf16* woutb = (__bf16*)(ws + (size_t)58720256);
  __bf16* qkvb  = (__bf16*)(ws + (size_t)67108864);
  __bf16* attnb = (__bf16*)(ws + (size_t)167772160);

  f32_to_bf16_kernel<<<4096, 256, 0, stream>>>(x,    xb,    16777216);
  f32_to_bf16_kernel<<<4096, 256, 0, stream>>>(Wqkv, wqkvb, 12582912);
  f32_to_bf16_kernel<<<2048, 256, 0, stream>>>(Wout, woutb, 4194304);

  // QKV: M=8192, N=6144, K=2048 -> (8192/32)*(6144/64) = 24576 waves / 8
  gemm_bf16_wmma<<<3072, 256, 0, stream>>>(xb, wqkvb, nullptr, qkvb,
                                           8192, 6144, 2048, 96);

  // Attention: B*H*(T/16) = 8192 waves / 4 per block
  attn_flash_wmma<<<2048, 128, 0, stream>>>(qkvb, attnb);

  // Output projection: M=8192, N=2048, K=2048 -> (8192/32)*(2048/64) = 8192 / 8
  gemm_bf16_wmma<<<1024, 256, 0, stream>>>(attnb, woutb, out, nullptr,
                                           8192, 2048, 2048, 32);
}
